// FasterRCNNDetector_80865644249476
// MI455X (gfx1250) — matrix operand order repaired and checked by reading, not compiled, not run on hardware
//
#include <hip/hip_runtime.h>
#include <hip/hip_bf16.h>
#include <cstdint>

// ---------------------------------------------------------------------------
// Problem constants (from the reference)
// ---------------------------------------------------------------------------
static constexpr int B_   = 4;
static constexpr int CF   = 256;
static constexpr int H_   = 128;
static constexpr int W_   = 128;
static constexpr int CLS  = 9;
static constexpr int K_   = 256;
static constexpr int OUT  = 7;
static constexpr int SR   = 2;
static constexpr int HW   = H_ * W_;
static constexpr int NROI = B_ * K_;              // 1024
static constexpr int NPIX = NROI * OUT * OUT;     // 50176 == GEMM M, divisible by 64
#define BN_EPS 1e-5f

// ---------------------------------------------------------------------------
// bf16 helpers + WMMA vector types (CDNA5 gfx1250, wave32)
// ---------------------------------------------------------------------------
typedef __attribute__((ext_vector_type(16))) __bf16 v16bf;
typedef __attribute__((ext_vector_type(8)))  float  v8f;

static __device__ __forceinline__ __bf16 f2bf(float f) {
  unsigned u = __builtin_bit_cast(unsigned, f);
  unsigned r = u + 0x7FFFu + ((u >> 16) & 1u);     // round to nearest even
  unsigned short s = (unsigned short)(r >> 16);
  return __builtin_bit_cast(__bf16, s);
}
static __device__ __forceinline__ float bf2f(__bf16 h) {
  unsigned short s = __builtin_bit_cast(unsigned short, h);
  unsigned u = ((unsigned)s) << 16;
  return __builtin_bit_cast(float, u);
}

// ---------------------------------------------------------------------------
// CDNA5 async DMA global->LDS (GLOBAL_LOAD_ASYNC_TO_LDS_B128, ASYNCcnt).
// VDST VGPR carries the wave-relative LDS byte offset (= low 32 bits of the
// generic pointer to a __shared__ object); VADDR pair carries the 64-bit
// global address (GV mode, saddr = off).
// ---------------------------------------------------------------------------
static __device__ __forceinline__ void async_load_b128(const void* gptr, void* lptr) {
  unsigned lds = (unsigned)(unsigned long long)(uintptr_t)lptr;
  unsigned long long ga = (unsigned long long)(uintptr_t)gptr;
  asm volatile("global_load_async_to_lds_b128 %0, %1, off"
               :: "v"(lds), "v"(ga)
               : "memory");
}
static __device__ __forceinline__ void wait_asynccnt0() {
  asm volatile("s_wait_asynccnt 0x0" ::: "memory");
}

// ---------------------------------------------------------------------------
// Kernel 1: decode top-K peaks into boxes (feature coords)
// ---------------------------------------------------------------------------
__global__ void bbox_kernel(const float* __restrict__ hm, const float* __restrict__ wh,
                            const float* __restrict__ off, const int* __restrict__ inds,
                            float* __restrict__ bbox) {
  int b = blockIdx.x, k = threadIdx.x;
  int ind = inds[b * K_ + k];
  const float* hmb = hm + (size_t)b * CLS * HW;
  float best = hmb[ind];
  int cls = 0;
  for (int c = 1; c < CLS; ++c) {
    float s = hmb[(size_t)c * HW + ind];
    if (s > best) { best = s; cls = c; }              // first-max, like jnp.argmax
  }
  const float* whb = wh + (size_t)b * CLS * 4 * HW + (size_t)cls * 4 * HW;
  float w0 = whb[0 * HW + ind], w1v = whb[1 * HW + ind];
  float w2v = whb[2 * HW + ind], w3v = whb[3 * HW + ind];
  const float* ofb = off + (size_t)b * 2 * HW;
  float xs = (float)(ind % W_) + ofb[ind];
  float ys = (float)(ind / W_) + ofb[HW + ind];
  float* o = bbox + (size_t)(b * K_ + k) * 4;
  o[0] = xs - w0; o[1] = ys - w1v; o[2] = xs + w2v; o[3] = ys + w3v;
}

// ---------------------------------------------------------------------------
// Kernel 2: ROI-align (bilinear, 2x2 avg) -> bf16 channel-last [NPIX, 256]
// One block per output pixel; 256 threads = 256 channels.
// ---------------------------------------------------------------------------
__global__ void roi_align_kernel(const float* __restrict__ x, const float* __restrict__ bbox,
                                 __bf16* __restrict__ rA) {
  int pixblk = blockIdx.x;                 // 0..NPIX-1
  int roi = pixblk / 49;
  int pix = pixblk % 49;
  int b   = roi / K_;
  int oy = pix / 7, ox = pix % 7;
  const float* bx = bbox + (size_t)roi * 4;
  float x1 = bx[0], y1 = bx[1], x2 = bx[2], y2 = bx[3];
  float rw = fmaxf(x2 - x1, 1.0f), rh = fmaxf(y2 - y1, 1.0f);
  const float inv_n = 1.0f / (float)(OUT * SR);     // 1/14

  int c = threadIdx.x;
  const float* f = x + ((size_t)b * CF + c) * HW;
  float acc = 0.f;
#pragma unroll
  for (int sy = 0; sy < SR; ++sy) {
#pragma unroll
    for (int sx = 0; sx < SR; ++sx) {
      float py = y1 + ((float)(oy * SR + sy) + 0.5f) * rh * inv_n;
      float px = x1 + ((float)(ox * SR + sx) + 0.5f) * rw * inv_n;
      py = fminf(fmaxf(py, 0.f), (float)(H_ - 1));
      px = fminf(fmaxf(px, 0.f), (float)(W_ - 1));
      int y0 = (int)floorf(py), x0 = (int)floorf(px);
      int y1i = min(y0 + 1, H_ - 1), x1i = min(x0 + 1, W_ - 1);
      float wy = py - (float)y0, wx = px - (float)x0;
      float v00 = f[y0 * W_ + x0],  v01 = f[y0 * W_ + x1i];
      float v10 = f[y1i * W_ + x0], v11 = f[y1i * W_ + x1i];
      acc += v00 * (1.f - wy) * (1.f - wx) + v01 * (1.f - wy) * wx +
             v10 * wy * (1.f - wx)         + v11 * wy * wx;
    }
  }
  rA[(size_t)pixblk * CF + c] = f2bf(acc * 0.25f);
}

// ---------------------------------------------------------------------------
// Kernel 3: convert conv weights to bf16 [K][N] (WMMA-B layout)
// Ranges: w1 256x128 | w2 1152x128 | w3 128x512 | wd 256x512
// ---------------------------------------------------------------------------
__global__ void prep_weights(const float* __restrict__ w1, const float* __restrict__ w2,
                             const float* __restrict__ w3, const float* __restrict__ wd,
                             __bf16* __restrict__ w1b, __bf16* __restrict__ w2b,
                             __bf16* __restrict__ w3b, __bf16* __restrict__ wdb) {
  int idx = blockIdx.x * blockDim.x + threadIdx.x;
  if (idx < 32768) {                                   // w1b[k][n] = w1[n][k]
    int k = idx >> 7, n = idx & 127;
    w1b[idx] = f2bf(w1[(size_t)n * 256 + k]);
  } else if (idx < 32768 + 147456) {                   // w2b[k'][n], k' = c*9 + ky*3 + kx
    int i = idx - 32768;
    int k = i >> 7, n = i & 127;
    int c = k / 9, t = k % 9;
    w2b[i] = f2bf(w2[((size_t)n * 128 + c) * 9 + t]);
  } else if (idx < 32768 + 147456 + 65536) {           // w3b[k][n] = w3[n][k]
    int i = idx - (32768 + 147456);
    int k = i >> 9, n = i & 511;
    w3b[i] = f2bf(w3[(size_t)n * 128 + k]);
  } else if (idx < 32768 + 147456 + 65536 + 131072) {  // wdb[k][n] = wd[n][k]
    int i = idx - (32768 + 147456 + 65536);
    int k = i >> 9, n = i & 511;
    wdb[i] = f2bf(wd[(size_t)n * 256 + k]);
  }
}

// ---------------------------------------------------------------------------
// Kernel 4: tiled bf16 WMMA GEMM with fused BN / residual / ReLU epilogue.
//   D[M,N] = epilogue( A[M,Kg] * Bm[Kg,N] )
// Block = 256 threads = 8 waves; block tile 64(M) x 128(N) x 32(K).
// Wave w: wm = w&1 -> 32-row slab, wn = w>>1 -> 32-col slab; each wave owns a
// 32x32 output patch = 2x2 WMMA tiles (4x v_wmma_f32_16x16x32_bf16 per K-step,
// A and B fragments each reused twice).
// Tiles are staged into LDS with GLOBAL_LOAD_ASYNC_TO_LDS_B128 (ASYNCcnt DMA);
// the implicit-im2col A path gathers through VGPRs + ds_store instead.
// MODE: 0 = BN+ReLU, 1 = BN only, 2 = BN + add(ident) + ReLU.
// ---------------------------------------------------------------------------
template <int MODE, bool IM2COL>
__global__ __launch_bounds__(256) void gemm_wmma(
    const __bf16* __restrict__ A, const __bf16* __restrict__ Bm,
    const float* __restrict__ g, const float* __restrict__ bb,
    const float* __restrict__ mu, const float* __restrict__ var,
    const __bf16* __restrict__ ident, __bf16* __restrict__ Dst,
    int Kg, int N) {
  __shared__ __bf16 As[64][32];    // [m][k]   4 KB
  __shared__ __bf16 Bs[32][128];   // [k][n]   8 KB

  const int tid  = threadIdx.x;
  const int lane = tid & 31;
  const int wave = tid >> 5;
  const int wm   = wave & 1;       // 0..1 -> 32-row slab
  const int wn   = wave >> 1;      // 0..3 -> 32-col slab
  const int m0   = blockIdx.x * 64;
  const int n0   = blockIdx.y * 128;

  v8f acc[2][2] = {};

  for (int kb = 0; kb < Kg; kb += 32) {
    __syncthreads();
    // ---- stage A tile (64x32) ----
    if (!IM2COL) {
      int row = tid >> 2, seg = tid & 3;               // 16B per thread, async DMA
      async_load_b128(A + (size_t)(m0 + row) * Kg + kb + seg * 8,
                      &As[row][seg * 8]);
    } else {
      int row = tid >> 2;
      int m = m0 + row;
      int roi = m / 49, pix = m % 49;
      int oy = pix / 7, ox = pix % 7;
#pragma unroll
      for (int j = 0; j < 8; ++j) {
        int kk = (tid & 3) * 8 + j;
        int kprime = kb + kk;
        int c = kprime / 9, t = kprime % 9;
        int ky = t / 3, kx = t % 3;
        int iy = oy + ky - 1, ix = ox + kx - 1;
        __bf16 v = __builtin_bit_cast(__bf16, (unsigned short)0);
        if (iy >= 0 && iy < 7 && ix >= 0 && ix < 7)
          v = A[(size_t)(roi * 49 + iy * 7 + ix) * 128 + c];
        As[row][kk] = v;
      }
    }
    // ---- stage B tile (32x128) via async DMA ----
    {
      int row = tid >> 3, seg = tid & 7;               // 32B per thread
      const __bf16* src = Bm + (size_t)(kb + row) * N + n0 + seg * 16;
      async_load_b128(src,     &Bs[row][seg * 16]);
      async_load_b128(src + 8, &Bs[row][seg * 16 + 8]);
    }
    wait_asynccnt0();          // DMA writes to LDS complete for this wave
    __syncthreads();           // (compiler also resolves DScnt for im2col path)

    // ---- A fragments: 16-bit A 16x32 layout (ISA 7.12.2) ----
    const int half = lane >> 4, r = lane & 15;
    v16bf af[2];
#pragma unroll
    for (int fm = 0; fm < 2; ++fm) {
#pragma unroll
      for (int i = 0; i < 8; ++i) {
        int bk = (i < 4) ? (half * 8 + 2 * i) : (16 + half * 8 + 2 * (i - 4));
        af[fm][2 * i]     = As[wm * 32 + fm * 16 + r][bk];
        af[fm][2 * i + 1] = As[wm * 32 + fm * 16 + r][bk + 1];
      }
    }
    // ---- B fragments: column n = lane%16, K in same order as A ----
    v16bf bf[2];
#pragma unroll
    for (int fn = 0; fn < 2; ++fn) {
#pragma unroll
      for (int v = 0; v < 8; ++v) {
        int bk = (v < 4) ? (half * 8 + 2 * v) : (16 + half * 8 + 2 * (v - 4));
        bf[fn][2 * v]     = Bs[bk][wn * 32 + fn * 16 + r];
        bf[fn][2 * v + 1] = Bs[bk + 1][wn * 32 + fn * 16 + r];
      }
    }
#pragma unroll
    for (int fm = 0; fm < 2; ++fm)
#pragma unroll
      for (int fn = 0; fn < 2; ++fn)
        acc[fm][fn] = __builtin_amdgcn_wmma_f32_16x16x32_bf16(
            false, af[fm], false, bf[fn], (short)0, acc[fm][fn], false, false);
  }

  // ---- fused BN (+residual) (+ReLU) epilogue; C/D layout per ISA 7.12.2 ----
  const int half = lane >> 4, r = lane & 15;
#pragma unroll
  for (int fn = 0; fn < 2; ++fn) {
    int n = n0 + wn * 32 + fn * 16 + r;
    float inv  = g[n] * rsqrtf(var[n] + BN_EPS);
    float beta = bb[n] - mu[n] * inv;
#pragma unroll
    for (int fm = 0; fm < 2; ++fm) {
#pragma unroll
      for (int j = 0; j < 8; ++j) {
        int m = m0 + wm * 32 + fm * 16 + half * 8 + j;
        float v = acc[fm][fn][j] * inv + beta;
        if (MODE == 2) v += bf2f(ident[(size_t)m * N + n]);
        if (MODE != 1) v = fmaxf(v, 0.f);
        Dst[(size_t)m * N + n] = f2bf(v);
      }
    }
  }
}

// ---------------------------------------------------------------------------
// Kernel 5: cls/reg 1x1 convs (N = 9 + 4, too small for WMMA tiles) -> d_out
// ---------------------------------------------------------------------------
__global__ void head_kernel(const __bf16* __restrict__ feat,
                            const float* __restrict__ wc, const float* __restrict__ bc,
                            const float* __restrict__ wr, const float* __restrict__ br,
                            float* __restrict__ out) {
  int idx = blockIdx.x * blockDim.x + threadIdx.x;
  if (idx >= NPIX * 13) return;
  int m = idx / 13, o = idx % 13;
  int roi = m / 49, pix = m % 49;
  const __bf16* f = feat + (size_t)m * 512;
  const float* wv;
  float acc;
  if (o < 9) { wv = wc + (size_t)o * 512; acc = bc[o]; }
  else       { wv = wr + (size_t)(o - 9) * 512; acc = br[o - 9]; }
  for (int c = 0; c < 512; ++c) acc += bf2f(f[c]) * wv[c];
  if (o < 9)
    out[((size_t)roi * 9 + o) * 49 + pix] = acc;
  else
    out[(size_t)NROI * 9 * 49 + ((size_t)roi * 4 + (o - 9)) * 49 + pix] = acc;
}

// ---------------------------------------------------------------------------
// Launch
// ---------------------------------------------------------------------------
extern "C" void kernel_launch(void* const* d_in, const int* in_sizes, int n_in,
                              void* d_out, int out_size, void* d_ws, size_t ws_size,
                              hipStream_t stream) {
  (void)in_sizes; (void)n_in; (void)out_size; (void)ws_size;

  const float* x      = (const float*)d_in[0];
  const float* hm     = (const float*)d_in[1];
  const float* wh     = (const float*)d_in[2];
  const float* offset = (const float*)d_in[3];
  const int*   inds   = (const int*)d_in[4];
  const float* w1 = (const float*)d_in[5];
  const float *g1 = (const float*)d_in[6], *b1 = (const float*)d_in[7],
              *m1 = (const float*)d_in[8], *v1 = (const float*)d_in[9];
  const float* w2 = (const float*)d_in[10];
  const float *g2 = (const float*)d_in[11], *b2 = (const float*)d_in[12],
              *m2 = (const float*)d_in[13], *v2 = (const float*)d_in[14];
  const float* w3 = (const float*)d_in[15];
  const float *g3 = (const float*)d_in[16], *b3 = (const float*)d_in[17],
              *m3 = (const float*)d_in[18], *v3 = (const float*)d_in[19];
  const float* wd = (const float*)d_in[20];
  const float *gd = (const float*)d_in[21], *bd = (const float*)d_in[22],
              *md = (const float*)d_in[23], *vd = (const float*)d_in[24];
  const float *wc = (const float*)d_in[25], *bc = (const float*)d_in[26];
  const float *wr = (const float*)d_in[27], *br = (const float*)d_in[28];

  char* base = (char*)d_ws;
  size_t off = 0;
  auto alloc = [&](size_t bytes) -> void* {
    void* p = base + off;
    off = (off + bytes + 255) & ~(size_t)255;
    return p;
  };
  float*  bbox   = (float*) alloc((size_t)NROI * 4 * sizeof(float));
  __bf16* rA     = (__bf16*)alloc((size_t)NPIX * 256 * 2);   // ROI feats, channel-last
  __bf16* w1b    = (__bf16*)alloc((size_t)256  * 128 * 2);
  __bf16* w2b    = (__bf16*)alloc((size_t)1152 * 128 * 2);
  __bf16* w3b    = (__bf16*)alloc((size_t)128  * 512 * 2);
  __bf16* wdb    = (__bf16*)alloc((size_t)256  * 512 * 2);
  __bf16* out1   = (__bf16*)alloc((size_t)NPIX * 128 * 2);
  __bf16* out2   = (__bf16*)alloc((size_t)NPIX * 128 * 2);
  __bf16* identb = (__bf16*)alloc((size_t)NPIX * 512 * 2);
  __bf16* featb  = (__bf16*)alloc((size_t)NPIX * 512 * 2);

  bbox_kernel<<<B_, K_, 0, stream>>>(hm, wh, offset, inds, bbox);
  prep_weights<<<(376832 + 255) / 256, 256, 0, stream>>>(w1, w2, w3, wd,
                                                         w1b, w2b, w3b, wdb);
  roi_align_kernel<<<NPIX, 256, 0, stream>>>(x, bbox, rA);

  dim3 blk(256);
  // conv1 1x1: 256->128, BN+ReLU
  gemm_wmma<0, false><<<dim3(NPIX / 64, 128 / 128), blk, 0, stream>>>(
      rA, w1b, g1, b1, m1, v1, nullptr, out1, 256, 128);
  // downsample 1x1: 256->512, BN only (residual identity branch)
  gemm_wmma<1, false><<<dim3(NPIX / 64, 512 / 128), blk, 0, stream>>>(
      rA, wdb, gd, bd, md, vd, nullptr, identb, 256, 512);
  // conv2 3x3 (implicit im2col in LDS staging): 128->128, BN+ReLU
  gemm_wmma<0, true><<<dim3(NPIX / 64, 128 / 128), blk, 0, stream>>>(
      out1, w2b, g2, b2, m2, v2, nullptr, out2, 1152, 128);
  // conv3 1x1: 128->512, BN + add identity + ReLU -> feat
  gemm_wmma<2, false><<<dim3(NPIX / 64, 512 / 128), blk, 0, stream>>>(
      out2, w3b, g3, b3, m3, v3, identb, featb, 128, 512);

  head_kernel<<<(NPIX * 13 + 255) / 256, 256, 0, stream>>>(featb, wc, bc, wr, br,
                                                           (float*)d_out);
}
